// EncoderLayer_76338748719484
// MI455X (gfx1250) — matrix operand order repaired and checked by reading, not compiled, not run on hardware
//
#include <hip/hip_runtime.h>

#define B_  16
#define S_  1024
#define D_  256
#define H_  8
#define F_  1024
#define HD_ 32

typedef __bf16 bf16;
typedef __attribute__((ext_vector_type(16))) __bf16 v16bf;
typedef __attribute__((ext_vector_type(8)))  __bf16 v8bf;
typedef __attribute__((ext_vector_type(8)))  float  v8f;

// ---------------------------------------------------------------------------
// WMMA helpers (layouts per CDNA5 ISA §7.12.2, wave32)
// ---------------------------------------------------------------------------

// A fragment, 16x32 bf16, row-major source with leading dim ld (elements).
// lane L: m = L&15, half = L>>4; K = {half*8 .. half*8+7} ∪ {16+half*8 .. +7}
__device__ __forceinline__ v16bf load_a16x32(const bf16* base, int ld, int lane) {
  int row  = lane & 15;
  int half = lane >> 4;
  const bf16* p = base + (size_t)row * ld + half * 8;
  v8bf lo = *(const v8bf*)p;          // K = half*8 .. half*8+7      (16B aligned)
  v8bf hi = *(const v8bf*)(p + 16);   // K = 16+half*8 .. +7         (16B aligned)
  v16bf r;
#pragma unroll
  for (int i = 0; i < 8; ++i) { r[i] = lo[i]; r[i + 8] = hi[i]; }
  return r;
}

// B fragment, 32x16 bf16 where B[k][n] = W[n][k], W row-major, leading dim ld.
// lane L: n = L&15, half = L>>4; K = {half*8 .. half*8+7} ∪ {16+half*8 .. +7}
__device__ __forceinline__ v16bf load_b32x16(const bf16* base, int ld, int lane) {
  int row  = lane & 15;
  int half = lane >> 4;
  const bf16* p = base + (size_t)row * ld + half * 8;
  v8bf lo = *(const v8bf*)p;
  v8bf hi = *(const v8bf*)(p + 16);
  v16bf r;
#pragma unroll
  for (int i = 0; i < 8; ++i) { r[i] = lo[i]; r[i + 8] = hi[i]; }
  return r;
}

__device__ __forceinline__ v8f wmma_bf16(v16bf a, v16bf b, v8f c) {
  return __builtin_amdgcn_wmma_f32_16x16x32_bf16(false, a, false, b,
                                                 (short)0, c, false, false);
}

// ---------------------------------------------------------------------------
// Elementwise / reduction kernels
// ---------------------------------------------------------------------------

__global__ __launch_bounds__(256) void cvt_bf16_kernel(const float* in, bf16* out, int n) {
  int i = blockIdx.x * 256 + threadIdx.x;
  if (i < n) out[i] = (bf16)in[i];
}

// per-batch mean/rstd over S*D elements (joint 2D layernorm)
__global__ __launch_bounds__(256) void ln_stats_kernel(const float* x, float* stats, int per) {
  int b = blockIdx.x;
  const float* p = x + (size_t)b * per;
  float s = 0.f, ss = 0.f;
  for (int i = threadIdx.x; i < per; i += 256) { float v = p[i]; s += v; ss += v * v; }
  __shared__ float sh[512];
  sh[threadIdx.x] = s;
  sh[256 + threadIdx.x] = ss;
  __syncthreads();
  for (int off = 128; off > 0; off >>= 1) {
    if ((int)threadIdx.x < off) {
      sh[threadIdx.x] += sh[threadIdx.x + off];
      sh[256 + threadIdx.x] += sh[256 + threadIdx.x + off];
    }
    __syncthreads();
  }
  if (threadIdx.x == 0) {
    float mean = sh[0] / (float)per;
    float var  = sh[256] / (float)per - mean * mean;
    stats[2 * b]     = mean;
    stats[2 * b + 1] = rsqrtf(var + 1e-5f);
  }
}

__global__ __launch_bounds__(256) void ln_apply_kernel(const float* x, const float* gamma,
                                                       const float* beta, const float* stats,
                                                       bf16* out, int SD) {
  int i = blockIdx.x * 256 + threadIdx.x;
  int b = i / SD;
  int sd = i - b * SD;
  float mean = stats[2 * b], rstd = stats[2 * b + 1];
  out[i] = (bf16)((x[i] - mean) * rstd * gamma[sd] + beta[sd]);
}

// ---------------------------------------------------------------------------
// Generic wave-tile GEMM:  out[m,n] = act( A[m,:] . W[n,:] + bias[n] ) + res[m,n]
// A: [M,K] bf16 row-major, W: [N,K] bf16 row-major. One 16x16 tile per wave.
// ---------------------------------------------------------------------------
__global__ __launch_bounds__(256) void gemm_kernel(const bf16* __restrict__ A,
                                                   const bf16* __restrict__ W,
                                                   const float* __restrict__ bias,
                                                   const float* __restrict__ res,
                                                   float* __restrict__ outF,
                                                   bf16* __restrict__ outB,
                                                   int M, int N, int K, int relu) {
  int lane = threadIdx.x & 31;
  int wid  = threadIdx.x >> 5;
  int tilesN = N >> 4;
  int tiles  = (M >> 4) * tilesN;
  int id = blockIdx.x * 8 + wid;          // wave-uniform
  if (id >= tiles) return;
  int tm = id / tilesN;
  int tn = id - tm * tilesN;

  const bf16* Abase = A + (size_t)(tm * 16) * K;
  const bf16* Wbase = W + (size_t)(tn * 16) * K;

  v8f acc = {};
  for (int k0 = 0; k0 < K; k0 += 32) {
    v16bf a = load_a16x32(Abase + k0, K, lane);
    v16bf b = load_b32x16(Wbase + k0, K, lane);
    acc = wmma_bf16(a, b, acc);
  }

  int half = lane >> 4;
  int nn   = lane & 15;
  int n    = tn * 16 + nn;
  float bv = bias ? bias[n] : 0.f;
#pragma unroll
  for (int r = 0; r < 8; ++r) {
    int m = tm * 16 + r + 8 * half;
    size_t idx = (size_t)m * N + n;
    float v = acc[r] + bv;
    if (relu) v = fmaxf(v, 0.f);
    if (res)  v += res[idx];
    if (outF) outF[idx] = v;
    if (outB) outB[idx] = (bf16)v;
  }
}

// ---------------------------------------------------------------------------
// Flash-style attention. qkv: [B*S, 768] bf16, head h uses cols [96h,96h+96)
// (q:+0..31, k:+32..63, v:+64..95).  One wave = one (b,h,16-query tile).
// Output stored pre-scrambled per the reference reshape:
//   attnv[b, h*128 + s/8, (s%8)*32 + hd]
// ---------------------------------------------------------------------------
__global__ __launch_bounds__(256) void attn_kernel(const bf16* __restrict__ qkv,
                                                   bf16* __restrict__ attnv) {
  __shared__ __align__(16) bf16 plds[8][16 * 32];   // per-wave P staging (1KB)

  int lane = threadIdx.x & 31;
  int wid  = threadIdx.x >> 5;
  int id = blockIdx.x * 8 + wid;        // 0 .. B*H*(S/16)-1 (exact)
  int qt = id & 63;
  int h  = (id >> 6) & 7;
  int b  = id >> 9;
  int s0 = qt * 16;
  int half = lane >> 4;
  int nn   = lane & 15;

  const bf16* rowbase = qkv + (size_t)b * S_ * 768;

  // Q fragment: 16 query rows x HD=32 (exactly one WMMA K)
  v16bf aq = load_a16x32(rowbase + (size_t)s0 * 768 + h * 96, 768, lane);

  float mi[8], li[8];
#pragma unroll
  for (int r = 0; r < 8; ++r) { mi[r] = -1e30f; li[r] = 0.f; }
  v8f o0 = {}, o1 = {};
  const float scale = 0.17677669529663687f;   // 1/sqrt(32)

  for (int kk = 0; kk < S_; kk += 32) {
    // ---- scores: S[m, kk+n] = q . k   (K^T fragments, two 16B row chunks/lane)
    const bf16* kbase = rowbase + (size_t)kk * 768 + h * 96 + 32;
    v16bf bk0 = load_b32x16(kbase, 768, lane);
    v16bf bk1 = load_b32x16(kbase + (size_t)16 * 768, 768, lane);
    v8f zero = {};
    v8f sv0 = wmma_bf16(aq, bk0, zero);
    v8f sv1 = wmma_bf16(aq, bk1, zero);

    // ---- online softmax over the 32 new keys (row reductions in 16-lane halves)
#pragma unroll
    for (int r = 0; r < 8; ++r) {
      float x0 = sv0[r] * scale, x1 = sv1[r] * scale;
      float rmax = fmaxf(x0, x1);
#pragma unroll
      for (int mm = 1; mm < 16; mm <<= 1) rmax = fmaxf(rmax, __shfl_xor(rmax, mm, 32));
      float mnew = fmaxf(mi[r], rmax);
      float c  = __expf(mi[r] - mnew);
      float p0 = __expf(x0 - mnew);
      float p1 = __expf(x1 - mnew);
      float rs = p0 + p1;
#pragma unroll
      for (int mm = 1; mm < 16; mm <<= 1) rs += __shfl_xor(rs, mm, 32);
      li[r] = li[r] * c + rs;
      mi[r] = mnew;
      o0[r] *= c;
      o1[r] *= c;
      int row = r + 8 * half;                    // C-tile row this lane owns
      plds[wid][row * 32 + nn]      = (bf16)p0;  // P transpose via LDS
      plds[wid][row * 32 + 16 + nn] = (bf16)p1;
    }
    asm volatile("" ::: "memory");               // keep LDS store->load order
    v16bf ap = load_a16x32(&plds[wid][0], 32, lane);
    asm volatile("" ::: "memory");

    // ---- O += P(16x32) @ V(32x32): V fragments gathered (strided rows)
    const bf16* vbase = rowbase + (size_t)kk * 768 + h * 96 + 64;
    int kb = half * 8;
    v16bf bv0, bv1;
#pragma unroll
    for (int i = 0; i < 16; ++i) {
      const bf16* vp = vbase + (size_t)(kb + i + (i >= 8 ? 8 : 0)) * 768 + nn;
      bv0[i] = vp[0];
      bv1[i] = vp[16];
    }
    o0 = wmma_bf16(ap, bv0, o0);
    o1 = wmma_bf16(ap, bv1, o1);
  }

  // ---- finalize + scrambled store (reference's permute-free reshape)
#pragma unroll
  for (int r = 0; r < 8; ++r) {
    int srow = s0 + r + 8 * half;
    float inv = 1.0f / li[r];
    int i = h * 128 + (srow >> 3);
    int jb = (srow & 7) << 5;
    size_t rowoff = ((size_t)b * 1024 + i) * 256 + jb;
    attnv[rowoff + nn]      = (bf16)(o0[r] * inv);
    attnv[rowoff + 16 + nn] = (bf16)(o1[r] * inv);
  }
}

// ---------------------------------------------------------------------------
// Launch
// ---------------------------------------------------------------------------
extern "C" void kernel_launch(void* const* d_in, const int* in_sizes, int n_in,
                              void* d_out, int out_size, void* d_ws, size_t ws_size,
                              hipStream_t stream) {
  const float* x      = (const float*)d_in[0];
  const float* gamma1 = (const float*)d_in[1];
  const float* beta1  = (const float*)d_in[2];
  const float* Wqkv   = (const float*)d_in[3];
  const float* bqkv   = (const float*)d_in[4];
  const float* Wo     = (const float*)d_in[5];
  const float* bo     = (const float*)d_in[6];
  const float* gamma2 = (const float*)d_in[7];
  const float* beta2  = (const float*)d_in[8];
  const float* W1     = (const float*)d_in[9];
  const float* b1     = (const float*)d_in[10];
  const float* W2     = (const float*)d_in[11];
  const float* b2     = (const float*)d_in[12];
  float* out = (float*)d_out;
  (void)in_sizes; (void)n_in; (void)out_size; (void)ws_size;

  char* ws = (char*)d_ws;
  size_t off = 0;
  auto carve = [&](size_t bytes) -> char* {
    char* p = ws + off;
    off += (bytes + 255) & ~(size_t)255;
    return p;
  };
  float* stats  = (float*)carve(2 * B_ * sizeof(float));             // reused LN1/LN2
  bf16* wqkv_bf = (bf16*)carve((size_t)3 * D_ * D_ * 2);
  bf16* wo_bf   = (bf16*)carve((size_t)D_ * D_ * 2);
  bf16* w1_bf   = (bf16*)carve((size_t)F_ * D_ * 2);
  bf16* w2_bf   = (bf16*)carve((size_t)D_ * F_ * 2);
  bf16* h_bf    = (bf16*)carve((size_t)B_ * S_ * D_ * 2);            // LN out (reused)
  bf16* qkv_bf  = (bf16*)carve((size_t)B_ * S_ * 3 * D_ * 2);
  bf16* attn_bf = (bf16*)carve((size_t)B_ * S_ * D_ * 2);
  bf16* t_bf    = (bf16*)carve((size_t)B_ * S_ * F_ * 2);

  const int MS = B_ * S_;   // 16384 rows
  const int SD = S_ * D_;   // 262144 per-batch LN span

  // weights -> bf16
  cvt_bf16_kernel<<<(3 * D_ * D_ + 255) / 256, 256, 0, stream>>>(Wqkv, wqkv_bf, 3 * D_ * D_);
  cvt_bf16_kernel<<<(D_ * D_ + 255) / 256, 256, 0, stream>>>(Wo, wo_bf, D_ * D_);
  cvt_bf16_kernel<<<(F_ * D_ + 255) / 256, 256, 0, stream>>>(W1, w1_bf, F_ * D_);
  cvt_bf16_kernel<<<(D_ * F_ + 255) / 256, 256, 0, stream>>>(W2, w2_bf, D_ * F_);

  // LN1 -> h_bf
  ln_stats_kernel<<<B_, 256, 0, stream>>>(x, stats, SD);
  ln_apply_kernel<<<(MS * D_) / 256, 256, 0, stream>>>(x, gamma1, beta1, stats, h_bf, SD);

  // QKV GEMM: [MS,256] x [768,256]^T -> qkv_bf
  gemm_kernel<<<(MS / 16) * (3 * D_ / 16) / 8, 256, 0, stream>>>(
      h_bf, wqkv_bf, bqkv, nullptr, nullptr, qkv_bf, MS, 3 * D_, D_, 0);

  // attention (writes scrambled-layout attn_bf)
  attn_kernel<<<(B_ * H_ * (S_ / 16)) / 8, 256, 0, stream>>>(qkv_bf, attn_bf);

  // Wo GEMM + residual(x) -> y1 in d_out
  gemm_kernel<<<(MS / 16) * (D_ / 16) / 8, 256, 0, stream>>>(
      attn_bf, wo_bf, bo, x, out, nullptr, MS, D_, D_, 0);

  // LN2 on y1 -> h_bf (reuse)
  ln_stats_kernel<<<B_, 256, 0, stream>>>(out, stats, SD);
  ln_apply_kernel<<<(MS * D_) / 256, 256, 0, stream>>>(out, gamma2, beta2, stats, h_bf, SD);

  // MLP: relu(h @ W1^T + b1) -> t_bf ; then t @ W2^T + b2 + y1 -> out
  gemm_kernel<<<(MS / 16) * (F_ / 16) / 8, 256, 0, stream>>>(
      h_bf, w1_bf, b1, nullptr, nullptr, t_bf, MS, F_, D_, 1);
  gemm_kernel<<<(MS / 16) * (D_ / 16) / 8, 256, 0, stream>>>(
      t_bf, w2_bf, b2, out, out, nullptr, MS, D_, F_, 0);
}
